// DeformConv_38835094290516
// MI455X (gfx1250) — matrix operand order repaired and checked by reading, not compiled, not run on hardware
//
#include <hip/hip_runtime.h>
#include <hip/hip_bf16.h>

// Problem constants (match reference)
#define B_    4
#define C_    256
#define H_    64
#define W_    64
#define O_    256
#define K_    9          // 3x3 taps
#define P_    (H_ * W_)  // 4096 pixels per batch
#define CK_   (C_ * K_)  // 2304 GEMM K-dim
#define OFFC  (2 * K_)   // 18 offset channels
#define OFFM  32         // offset-GEMM M padded to 2 WMMA tiles
#define EPS_  1e-5f

typedef __attribute__((ext_vector_type(16))) __bf16 v16bf;
typedef __attribute__((ext_vector_type(8)))  __bf16 v8bf;
typedef __attribute__((ext_vector_type(8)))  float  v8f;

// ---------------------------------------------------------------------------
// WMMA helpers (A-fragment layout: lane m=lane%16, 16 contiguous K at
// kb = (lane/16)*16; B-fragment: 16 contiguous K at
// K-half (lane/16)*16, column n = lane%16)
// ---------------------------------------------------------------------------
__device__ __forceinline__ v16bf load_a_frag(const __bf16* row, int k0, int kbA) {
    return *(const v16bf*)(row + k0 + kbA);
}

__device__ __forceinline__ v8f wmma_bf16(v16bf a, v16bf b, v8f c) {
    return __builtin_amdgcn_wmma_f32_16x16x32_bf16(false, a, false, b,
                                                   (short)0, c, false, false);
}

// ---------------------------------------------------------------------------
// 1) Regular (zero-padded) im2col in bf16: R[b*P+p][k*C+c]
//    One wave per (b,p,k); lane owns c = lane*8..lane*8+7 -> one v8bf store;
//    a wave writes one contiguous 512B burst.
// ---------------------------------------------------------------------------
__global__ void __launch_bounds__(256)
im2col_kernel(const float* __restrict__ x,   // [B][C][H][W]
              __bf16* __restrict__ R)        // [B*P][CK]
{
    int tid   = blockIdx.x * blockDim.x + threadIdx.x;
    int chunk = tid & 31;          // 8-channel chunk within (b,p,k)
    int idx   = tid >> 5;          // (b, p, k)
    if (idx >= B_ * P_ * K_) return;
    int k = idx % K_;
    int p = (idx / K_) % P_;
    int b = idx / (K_ * P_);
    int px = p % W_;
    int py = p / W_;
    int iy = py - 1 + k / 3;
    int ix = px - 1 + k % 3;

    v8bf outv;
    if (iy >= 0 && iy < H_ && ix >= 0 && ix < W_) {
        const float* xc = x + (size_t)b * C_ * P_
                            + (size_t)(chunk * 8) * P_ + iy * W_ + ix;
        #pragma unroll
        for (int cc = 0; cc < 8; ++cc)
            outv[cc] = (__bf16)xc[(size_t)cc * P_];
    } else {
        #pragma unroll
        for (int cc = 0; cc < 8; ++cc)
            outv[cc] = (__bf16)0.f;
    }
    *(v8bf*)(R + ((size_t)(b * P_ + p)) * CK_ + k * C_ + chunk * 8) = outv;
}

// ---------------------------------------------------------------------------
// 2) Offset-conv weights -> bf16 [32][k*C+c] (rows 18..31 zero-padded)
// ---------------------------------------------------------------------------
__global__ void __launch_bounds__(256)
prep_off_kernel(const float* __restrict__ w,   // [18][C][3][3]
                __bf16* __restrict__ Woff)     // [32][CK]
{
    for (int e = blockIdx.x * blockDim.x + threadIdx.x;
         e < OFFM * CK_; e += gridDim.x * blockDim.x) {
        int o = e / CK_;
        int r = e % CK_;
        int k = r / C_;
        int c = r % C_;
        float v = (o < OFFC) ? w[((size_t)o * C_ + c) * 9 + k] : 0.f;
        Woff[e] = (__bf16)v;
    }
}

// ---------------------------------------------------------------------------
// 3) Offset conv as WMMA GEMM: off[b][oc][p] = b_off[oc] + Woff x R
//    One wave: M=32 (all channels, padded), N=64.  Waves = B * P/64 = 256.
// ---------------------------------------------------------------------------
__global__ void __launch_bounds__(256)
offset_gemm_kernel(const __bf16* __restrict__ Woff,  // [32][CK]
                   const __bf16* __restrict__ R,     // [B*P][CK]
                   const float* __restrict__ bo,     // [18]
                   float* __restrict__ off)          // [B][18][P]
{
    const int lane = threadIdx.x & 31;
    const int wid  = blockIdx.x * (blockDim.x >> 5) + (threadIdx.x >> 5);
    const int ntile_p = P_ / 64;
    int pg = wid % ntile_p;
    int b  = wid / ntile_p;
    if (b >= B_) return;

    const int m    = lane & 15;
    const int half = lane >> 4;
    const int kbA  = half * 16;
    const int ksB  = half * 16;

    const __bf16* A0 = Woff + (size_t)m * CK_;
    const __bf16* A1 = A0 + (size_t)16 * CK_;
    const __bf16* Br = R + ((size_t)(b * P_ + pg * 64 + m)) * CK_;

    v8f acc[2][4] = {};
    for (int k0 = 0; k0 < CK_; k0 += 32) {
        v16bf af[2];
        af[0] = load_a_frag(A0, k0, kbA);
        af[1] = load_a_frag(A1, k0, kbA);
        v16bf bf[4];
        #pragma unroll
        for (int i = 0; i < 4; ++i)
            bf[i] = *(const v16bf*)(Br + (size_t)i * 16 * CK_ + k0 + ksB);
        #pragma unroll
        for (int j = 0; j < 2; ++j)
            #pragma unroll
            for (int i = 0; i < 4; ++i)
                acc[j][i] = wmma_bf16(af[j], bf[i], acc[j][i]);
    }

    const int n  = lane & 15;
    const int p0 = pg * 64;
    #pragma unroll
    for (int j = 0; j < 2; ++j)
        #pragma unroll
        for (int v = 0; v < 8; ++v) {
            int o = j * 16 + v + 8 * half;
            if (o < OFFC) {
                float bb = bo[o];
                size_t base = ((size_t)b * OFFC + o) * P_ + p0 + n;
                #pragma unroll
                for (int i = 0; i < 4; ++i)
                    off[base + i * 16] = acc[j][i][v] + bb;
            }
        }
}

// ---------------------------------------------------------------------------
// 4) Bilinear sampling fused with bf16 im2col write: S[b*P+p][k*C+c]
//    (S aliases R's storage; offset GEMM has completed by then)
//    One wave per (b,p,k); lane owns an 8-channel chunk -> one v8bf store.
// ---------------------------------------------------------------------------
__global__ void __launch_bounds__(256)
sample_kernel(const float* __restrict__ x,
              const float* __restrict__ off,
              __bf16* __restrict__ S)
{
    int tid   = blockIdx.x * blockDim.x + threadIdx.x;
    int chunk = tid & 31;          // 8-channel chunk within (b,p,k)
    int idx   = tid >> 5;          // (b, p, k)
    if (idx >= B_ * P_ * K_) return;
    int k = idx % K_;
    int p = (idx / K_) % P_;
    int b = idx / (K_ * P_);
    int px = p % W_;
    int py = p / W_;

    float dy = off[((size_t)b * OFFC + 2 * k)     * P_ + p];
    float dx = off[((size_t)b * OFFC + 2 * k + 1) * P_ + p];
    float sy = (float)(py - 1 + k / 3) + dy;
    float sx = (float)(px - 1 + k % 3) + dx;

    float y0f = floorf(sy), x0f = floorf(sx);
    float ly = sy - y0f, lx = sx - x0f;
    int y0 = (int)y0f, x0 = (int)x0f;
    int y1 = y0 + 1,   x1 = x0 + 1;

    float vy0 = (y0 >= 0 && y0 < H_) ? 1.f : 0.f;
    float vy1 = (y1 >= 0 && y1 < H_) ? 1.f : 0.f;
    float vx0 = (x0 >= 0 && x0 < W_) ? 1.f : 0.f;
    float vx1 = (x1 >= 0 && x1 < W_) ? 1.f : 0.f;
    float w00 = (1.f - ly) * (1.f - lx) * vy0 * vx0;
    float w01 = (1.f - ly) * lx         * vy0 * vx1;
    float w10 = ly * (1.f - lx)         * vy1 * vx0;
    float w11 = ly * lx                 * vy1 * vx1;

    int y0c = min(max(y0, 0), H_ - 1);
    int y1c = min(max(y1, 0), H_ - 1);
    int x0c = min(max(x0, 0), W_ - 1);
    int x1c = min(max(x1, 0), W_ - 1);
    int i00 = y0c * W_ + x0c, i01 = y0c * W_ + x1c;
    int i10 = y1c * W_ + x0c, i11 = y1c * W_ + x1c;

    const float* xb = x + (size_t)b * C_ * P_ + (size_t)(chunk * 8) * P_;
    v8bf outv;
    #pragma unroll
    for (int cc = 0; cc < 8; ++cc) {
        const float* xc = xb + (size_t)cc * P_;
        float v = xc[i00] * w00 + xc[i01] * w01 + xc[i10] * w10 + xc[i11] * w11;
        outv[cc] = (__bf16)v;
    }
    *(v8bf*)(S + ((size_t)(b * P_ + p)) * CK_ + k * C_ + chunk * 8) = outv;
}

// ---------------------------------------------------------------------------
// 5) Main weights -> bf16 [o][k*C+c]; fold BN+bias into per-o scale/shift
// ---------------------------------------------------------------------------
__global__ void __launch_bounds__(256)
prep_kernel(const float* __restrict__ weight,   // [O][C][9]
            const float* __restrict__ bias,
            const float* __restrict__ gamma,
            const float* __restrict__ beta,
            const float* __restrict__ mean,
            const float* __restrict__ var,
            __bf16* __restrict__ Wb,            // [O][CK]
            float* __restrict__ sc,
            float* __restrict__ sh)
{
    int idx = blockIdx.x * blockDim.x + threadIdx.x;
    if (idx < O_) {
        float inv = gamma[idx] * rsqrtf(var[idx] + EPS_);
        sc[idx] = inv;
        sh[idx] = beta[idx] + (bias[idx] - mean[idx]) * inv;
    }
    for (int e = idx; e < O_ * CK_; e += gridDim.x * blockDim.x) {
        int o = e / CK_;
        int r = e % CK_;
        int k = r / C_;
        int c = r % C_;
        Wb[e] = (__bf16)weight[((size_t)o * C_ + c) * 9 + k];
    }
}

// ---------------------------------------------------------------------------
// 6) Main WMMA GEMM, register double-buffered.
//    Wave tile: M=32 (o) x N=64 (p); 8 WMMAs per K-step of 32.
//    out[b][o][p] = relu(sc[o]*acc + sh[o])
// ---------------------------------------------------------------------------
__global__ void __launch_bounds__(256)
dcn_gemm_kernel(const __bf16* __restrict__ Wb,  // [O][CK]
                const __bf16* __restrict__ S,   // [B*P][CK]
                const float* __restrict__ sc,
                const float* __restrict__ sh,
                float* __restrict__ out)        // [B][O][P]
{
    const int lane = threadIdx.x & 31;
    const int wid  = blockIdx.x * (blockDim.x >> 5) + (threadIdx.x >> 5);

    const int ntile_p = P_ / 64;   // 64
    const int ntile_o = O_ / 32;   // 8
    int t  = wid;
    int pg = t % ntile_p;  t /= ntile_p;
    int ot = t % ntile_o;  t /= ntile_o;
    int b  = t;
    if (b >= B_) return;

    const int m    = lane & 15;
    const int half = lane >> 4;
    const int kbA  = half * 16;
    const int ksB  = half * 16;

    const __bf16* A0 = Wb + ((size_t)(ot * 32 + m)) * CK_;
    const __bf16* A1 = A0 + (size_t)16 * CK_;
    const __bf16* Br = S  + ((size_t)(b * P_ + pg * 64 + m)) * CK_;

    v8f acc[2][4] = {};

    // ---- software pipeline: prologue ----
    v16bf af[2], bf[4];
    af[0] = load_a_frag(A0, 0, kbA);
    af[1] = load_a_frag(A1, 0, kbA);
    #pragma unroll
    for (int i = 0; i < 4; ++i)
        bf[i] = *(const v16bf*)(Br + (size_t)i * 16 * CK_ + ksB);

    for (int k0 = 0; k0 < CK_ - 32; k0 += 32) {
        const int kn = k0 + 32;
        __builtin_prefetch((const void*)(A0 + k0 + 256), 0, 3);
        __builtin_prefetch((const void*)(Br + k0 + 256), 0, 3);
        __builtin_prefetch((const void*)(Br + (size_t)2 * 16 * CK_ + k0 + 256), 0, 3);

        // issue next-iteration loads before consuming current fragments
        v16bf afn[2], bfn[4];
        afn[0] = load_a_frag(A0, kn, kbA);
        afn[1] = load_a_frag(A1, kn, kbA);
        #pragma unroll
        for (int i = 0; i < 4; ++i)
            bfn[i] = *(const v16bf*)(Br + (size_t)i * 16 * CK_ + kn + ksB);

        #pragma unroll
        for (int j = 0; j < 2; ++j)
            #pragma unroll
            for (int i = 0; i < 4; ++i)
                acc[j][i] = wmma_bf16(af[j], bf[i], acc[j][i]);

        af[0] = afn[0]; af[1] = afn[1];
        #pragma unroll
        for (int i = 0; i < 4; ++i) bf[i] = bfn[i];
    }
    // ---- epilogue of pipeline: last K-step ----
    #pragma unroll
    for (int j = 0; j < 2; ++j)
        #pragma unroll
        for (int i = 0; i < 4; ++i)
            acc[j][i] = wmma_bf16(af[j], bf[i], acc[j][i]);

    // ---- store: C/D layout -> col n = lane&15, row m = v + 8*(lane>>4) ----
    const int n  = lane & 15;
    const int p0 = pg * 64;
    float* outb = out + (size_t)b * O_ * P_;
    #pragma unroll
    for (int j = 0; j < 2; ++j)
        #pragma unroll
        for (int v = 0; v < 8; ++v) {
            int o = ot * 32 + j * 16 + v + 8 * half;
            float s  = sc[o];
            float t2 = sh[o];
            size_t base = (size_t)o * P_ + p0 + n;
            #pragma unroll
            for (int i = 0; i < 4; ++i)
                outb[base + i * 16] = fmaxf(fmaf(acc[j][i][v], s, t2), 0.f);
        }
}

// ---------------------------------------------------------------------------
// Launch
// ---------------------------------------------------------------------------
extern "C" void kernel_launch(void* const* d_in, const int* in_sizes, int n_in,
                              void* d_out, int out_size, void* d_ws, size_t ws_size,
                              hipStream_t stream) {
    const float* input_x  = (const float*)d_in[0];
    const float* w_offset = (const float*)d_in[1];
    const float* b_offset = (const float*)d_in[2];
    const float* weight   = (const float*)d_in[3];
    const float* bias     = (const float*)d_in[4];
    const float* gamma    = (const float*)d_in[5];
    const float* beta     = (const float*)d_in[6];
    const float* run_mean = (const float*)d_in[7];
    const float* run_var  = (const float*)d_in[8];
    float* out = (float*)d_out;

    // Workspace carve-up (segments 256B aligned)
    char* ws = (char*)d_ws;
    float*  off  = (float*)ws;                                  // B*18*P f32
    size_t  o1   = (size_t)B_ * OFFC * P_ * sizeof(float);      // 1,179,648
    __bf16* Wb   = (__bf16*)(ws + o1);                          // O*CK bf16
    size_t  o2   = o1 + (size_t)O_ * CK_ * sizeof(__bf16);      // +1,179,648
    __bf16* Woff = (__bf16*)(ws + o2);                          // 32*CK bf16
    size_t  o3   = o2 + (size_t)OFFM * CK_ * sizeof(__bf16);    // +147,456
    float*  sc   = (float*)(ws + o3);                           // O f32
    float*  sh   = sc + O_;                                     // O f32
    size_t  o4   = (o3 + 2 * O_ * sizeof(float) + 255) & ~(size_t)255;
    __bf16* Sbuf = (__bf16*)(ws + o4);   // B*P*CK bf16 (~75.5MB); holds R then S

    {   // 1) regular im2col (R), one wave per (b,p,k)
        long long nthr = (long long)B_ * P_ * K_ * 32;
        im2col_kernel<<<(int)((nthr + 255) / 256), 256, 0, stream>>>(input_x, Sbuf);
    }
    {   // 2) offset weights -> bf16 (padded to 32 rows)
        prep_off_kernel<<<256, 256, 0, stream>>>(w_offset, Woff);
    }
    {   // 3) offset conv on the matrix pipes
        int waves  = B_ * (P_ / 64);       // 256 waves
        int blocks = waves / 8;
        offset_gemm_kernel<<<blocks, 256, 0, stream>>>(Woff, Sbuf, b_offset, off);
    }
    {   // 4) deformed bilinear sample -> S (overwrites R storage)
        long long nthr = (long long)B_ * P_ * K_ * 32;
        sample_kernel<<<(int)((nthr + 255) / 256), 256, 0, stream>>>(input_x, off, Sbuf);
    }
    {   // 5) main weight convert + BN fold
        prep_kernel<<<1024, 256, 0, stream>>>(
            weight, bias, gamma, beta, run_mean, run_var, Wb, sc, sh);
    }
    {   // 6) main WMMA GEMM (M=32 x N=64 per wave, double-buffered)
        int waves  = B_ * (O_ / 32) * (P_ / 64);   // 2048 waves
        int blocks = waves / 8;                    // 256 blocks of 8 waves
        dcn_gemm_kernel<<<blocks, 256, 0, stream>>>(Wb, Sbuf, sc, sh, out);
    }
}
